// MMM_65111704207907
// MI455X (gfx1250) — compile-verified, matrix operand order checked by reading, not run on hardware
//
#include <hip/hip_runtime.h>
#include <hip/hip_bf16.h>

typedef _Float16 half16 __attribute__((ext_vector_type(16)));
typedef float    float8 __attribute__((ext_vector_type(8)));
typedef unsigned uint4v __attribute__((ext_vector_type(4)));
typedef unsigned uint8v __attribute__((ext_vector_type(8)));

#define LOG2PI 1.8378770664093453f

// Workspace layout (bytes). Total ~8.67 MB.
#define B_OFF     0u        // packed B matrix: [512 cols][256 K] f16 = 256 KB
#define CONST_OFF 262144u   // per-column constant: [512] f32
#define LOGA_OFF  264192u   // log_A: [64][64] f32
#define LOGPI_OFF 280576u   // log_pi: [64] f32
#define EMIS_OFF  281088u   // emis: [8*4096][64] f32 = 8 MB

// LDS byte address of a generic pointer to __shared__.
__device__ __forceinline__ unsigned lds_addr_of(const void* p) {
    return (unsigned)(unsigned long long)p;
}

// ---------------------------------------------------------------------------
// Kernel 1: parameter preprocessing.
// ---------------------------------------------------------------------------
__global__ void __launch_bounds__(512)
hmm_precompute(const float* __restrict__ pi_logits,
               const float* __restrict__ trans_logits,
               const float* __restrict__ weight_logits,
               const float* __restrict__ means,
               const float* __restrict__ log_vars,
               char* __restrict__ ws)
{
    _Float16* Bcat = (_Float16*)(ws + B_OFF);
    float*    cst  = (float*)(ws + CONST_OFF);
    float*    lA   = (float*)(ws + LOGA_OFF);
    float*    lpi  = (float*)(ws + LOGPI_OFF);
    __shared__ float lse_s[64];

    const int tid = threadIdx.x;              // 0..511, one mixture column each
    {
        const int j = tid;                    // j = s*8 + m
        float sum_lv = 0.0f, c = 0.0f;
        for (int d = 0; d < 128; ++d) {
            float lv = log_vars[j * 128 + d];
            float iv = __expf(-lv);
            float mu = means[j * 128 + d];
            Bcat[j * 256 + d]       = (_Float16)(-0.5f * iv);
            Bcat[j * 256 + 128 + d] = (_Float16)(mu * iv);
            sum_lv += lv;
            c += mu * mu * iv;
        }
        cst[j] = -0.5f * (sum_lv + 128.0f * LOG2PI) - 0.5f * c;
    }
    __syncthreads();
    if (tid < 64) {                            // log-softmax denom over M=8
        float mx = -__builtin_inff();
        for (int m = 0; m < 8; ++m) mx = fmaxf(mx, weight_logits[tid * 8 + m]);
        float s = 0.0f;
        for (int m = 0; m < 8; ++m) s += __expf(weight_logits[tid * 8 + m] - mx);
        lse_s[tid] = mx + __logf(s);
    }
    __syncthreads();
    cst[tid] += weight_logits[tid] - lse_s[tid >> 3];

    if (tid < 64) {                            // log_A row tid
        float mx = -__builtin_inff();
        for (int j = 0; j < 64; ++j) mx = fmaxf(mx, trans_logits[tid * 64 + j]);
        float s = 0.0f;
        for (int j = 0; j < 64; ++j) s += __expf(trans_logits[tid * 64 + j] - mx);
        float l = mx + __logf(s);
        for (int j = 0; j < 64; ++j) lA[tid * 64 + j] = trans_logits[tid * 64 + j] - l;
    }
    if (tid == 0) {                            // log_pi
        float mx = -__builtin_inff();
        for (int j = 0; j < 64; ++j) mx = fmaxf(mx, pi_logits[j]);
        float s = 0.0f;
        for (int j = 0; j < 64; ++j) s += __expf(pi_logits[j] - mx);
        float l = mx + __logf(s);
        for (int j = 0; j < 64; ++j) lpi[j] = pi_logits[j] - l;
    }
}

// ---------------------------------------------------------------------------
// Kernel 2: fused emission GEMM (WMMA f16 -> f32 acc) + logsumexp over M.
//   A tile (32 rows x 128 f32 of X) staged into LDS by the Tensor Data Mover:
//   each wave issues one tensor_load_to_lds for a 4x128 2D tile, waits on
//   TENSORcnt, then converts to the [x^2 ; x] f16 WMMA A-layout.
// ---------------------------------------------------------------------------
__global__ void __launch_bounds__(256)
hmm_emission_wmma(const float* __restrict__ X, char* __restrict__ ws)
{
    const _Float16* Bcat = (const _Float16*)(ws + B_OFF);
    const float*    cst  = (const float*)(ws + CONST_OFF);
    float*          emis = (float*)(ws + EMIS_OFF);

    __shared__ float    Xraw[32 * 128];   // 16 KB raw tile (TDM destination)
    __shared__ _Float16 Alds[32 * 256];   // 16 KB A operand (f16, K=256)

    const int tid   = threadIdx.x;
    const int wgrow = blockIdx.x * 32;

    // ---- TDM stage: wave w DMAs rows [w*4, w*4+4) of the tile (2 KB each) ----
    {
        const int w = __builtin_amdgcn_readfirstlane(tid >> 5);   // wave id (SGPR)
        unsigned long long ga =
            (unsigned long long)(X + ((size_t)wgrow + (size_t)w * 4) * 128);
        unsigned la = lds_addr_of(&Xraw[w * 4 * 128]);

        uint4v g0;
        g0[0] = 1u;                                   // count=1, no gather
        g0[1] = la;                                   // lds_addr
        g0[2] = (unsigned)ga;                         // global_addr[31:0]
        g0[3] = (unsigned)((ga >> 32) & 0x01FFFFFFu)  // global_addr[56:32]
              | (2u << 30);                           // type = 2 (image)

        uint8v g1;
        g1[0] = (2u << 16);        // workgroup_mask=0, data_size=2 (4 bytes)
        g1[1] = (128u << 16);      // tensor_dim0[15:0]=128 in [31:16]
        g1[2] = (4u << 16);        // tensor_dim0[31:16]=0 ; tensor_dim1[15:0]=4
        g1[3] = (128u << 16);      // tensor_dim1[31:16]=0 ; tile_dim0=128
        g1[4] = 4u;                // tile_dim1=4 ; tile_dim2=0
        g1[5] = 128u;              // tensor_dim0_stride[31:0]=128
        g1[6] = 0u;                // stride[47:32]=0 ; tensor_dim1_stride[15:0]=0
        g1[7] = 0u;                // tensor_dim1_stride[47:16]=0

        asm volatile("tensor_load_to_lds %0, %1"
                     :
                     : "s"(g0), "s"(g1)
                     : "memory");
        __builtin_amdgcn_s_wait_tensorcnt(0);
    }
    __syncthreads();

    // ---- Convert raw f32 rows into f16 A-layout: row r = [x^2 (0..127), x] ----
    {
        const int r  = tid >> 3;
        const int d0 = (tid & 7) * 16;
        #pragma unroll 4
        for (int dd = 0; dd < 16; ++dd) {
            float x = Xraw[r * 128 + d0 + dd];
            Alds[r * 256 + d0 + dd]       = (_Float16)(x * x);
            Alds[r * 256 + 128 + d0 + dd] = (_Float16)x;
        }
    }
    __syncthreads();

    const unsigned* Aw = (const unsigned*)Alds;
    const int wave    = tid >> 5;
    const int lane    = tid & 31;
    const int colLane = lane & 15;
    const int hi      = lane >> 4;        // 0/1: lane half selects K sub-block

    for (int q = 0; q < 8; ++q) {
        const int id = wave * 8 + q;
        const int rt = id >> 5;           // row tile 0/1
        const int ct = id & 31;           // column tile 0..31
        const int jcol = ct * 16 + colLane;
        const unsigned* Bw = (const unsigned*)(Bcat + (size_t)jcol * 256);
        const int arow = rt * 16 + colLane;

        float8 acc = {0.f, 0.f, 0.f, 0.f, 0.f, 0.f, 0.f, 0.f};
        #pragma unroll
        for (int kk = 0; kk < 8; ++kk) {
            union { half16 v; unsigned u[8]; } a, b;
            #pragma unroll
            for (int i = 0; i < 8; ++i) {
                // 16-bit A 16x32 layout: K = (i>=4)*16 + (lane>=16)*8 + (i%4)*2
                int koff = kk * 32 + ((i & 4) << 2) + (hi << 3) + ((i & 3) << 1);
                a.u[i] = Aw[(arow * 256 + koff) >> 1];   // LDS pair load
                b.u[i] = Bw[koff >> 1];                  // global pair load (L2)
            }
            acc = __builtin_amdgcn_wmma_f32_16x16x32_f16(
                false, a.v, false, b.v, (short)0, acc, false, false);
        }

        // Epilogue: add per-column constant, logsumexp over M=8 (8-lane groups).
        const float cj = cst[jcol];
        #pragma unroll
        for (int r = 0; r < 8; ++r) {
            float v = acc[r] + cj;
            float m = v;
            m = fmaxf(m, __shfl_xor(m, 1, 32));
            m = fmaxf(m, __shfl_xor(m, 2, 32));
            m = fmaxf(m, __shfl_xor(m, 4, 32));
            float s = __expf(v - m);
            s += __shfl_xor(s, 1, 32);
            s += __shfl_xor(s, 2, 32);
            s += __shfl_xor(s, 4, 32);
            if ((lane & 7) == 0) {
                int row = wgrow + rt * 16 + r + (hi << 3);
                int st  = jcol >> 3;
                emis[(size_t)row * 64 + st] = m + __logf(s);
            }
        }
    }
}

// ---------------------------------------------------------------------------
// Kernel 3: HMM forward scan (log-semiring). One block per batch chain.
// ---------------------------------------------------------------------------
__global__ void __launch_bounds__(256)
hmm_scan(const char* __restrict__ ws, float* __restrict__ out)
{
    __shared__ float lA[64 * 64];       // 16 KB transition matrix
    __shared__ float buf0[64], buf1[64];

    const float* lAg  = (const float*)(ws + LOGA_OFF);
    const float* lpi  = (const float*)(ws + LOGPI_OFF);
    const float* emis = (const float*)(ws + EMIS_OFF);

    const int b   = blockIdx.x;
    const int tid = threadIdx.x;
    for (int i = tid; i < 4096; i += 256) lA[i] = lAg[i];
    if (tid < 64) buf0[tid] = lpi[tid] + emis[(size_t)b * 4096 * 64 + tid];
    __syncthreads();

    const int j = tid >> 2;             // destination state
    const int p = tid & 3;              // partial-reduction slot

    for (int t = 1; t < 4096; ++t) {
        const float* rd = (t & 1) ? buf0 : buf1;
        float*       wr = (t & 1) ? buf1 : buf0;

        float vmax = -__builtin_inff();
        #pragma unroll
        for (int ii = 0; ii < 16; ++ii) {
            int i = p * 16 + ii;
            vmax = fmaxf(vmax, rd[i] + lA[i * 64 + j]);
        }
        float s = 0.0f;
        #pragma unroll
        for (int ii = 0; ii < 16; ++ii) {
            int i = p * 16 + ii;
            s += __expf(rd[i] + lA[i * 64 + j] - vmax);
        }
        #pragma unroll
        for (int off = 1; off <= 2; off <<= 1) {
            float om = __shfl_xor(vmax, off, 32);
            float os = __shfl_xor(s,    off, 32);
            float nm = fmaxf(vmax, om);
            s = s * __expf(vmax - nm) + os * __expf(om - nm);
            vmax = nm;
        }
        float val = emis[((size_t)b * 4096 + t) * 64 + j] + vmax + __logf(s);
        if (p == 0) wr[j] = val;
        __syncthreads();
    }

    if (tid == 0) {                      // final logsumexp over states
        const float* fin = buf1;         // t=4095 (odd) wrote buf1
        float mx = -__builtin_inff();
        for (int i = 0; i < 64; ++i) mx = fmaxf(mx, fin[i]);
        float s = 0.0f;
        for (int i = 0; i < 64; ++i) s += __expf(fin[i] - mx);
        out[b] = mx + __logf(s);
    }
}

// ---------------------------------------------------------------------------
extern "C" void kernel_launch(void* const* d_in, const int* in_sizes, int n_in,
                              void* d_out, int out_size, void* d_ws, size_t ws_size,
                              hipStream_t stream)
{
    (void)in_sizes; (void)n_in; (void)out_size; (void)ws_size;
    const float* X  = (const float*)d_in[0];
    const float* pi = (const float*)d_in[1];
    const float* tr = (const float*)d_in[2];
    const float* wl = (const float*)d_in[3];
    const float* mu = (const float*)d_in[4];
    const float* lv = (const float*)d_in[5];
    char*  ws  = (char*)d_ws;
    float* out = (float*)d_out;

    hipLaunchKernelGGL(hmm_precompute,    dim3(1),    dim3(512), 0, stream,
                       pi, tr, wl, mu, lv, ws);
    hipLaunchKernelGGL(hmm_emission_wmma, dim3(1024), dim3(256), 0, stream,
                       X, ws);
    hipLaunchKernelGGL(hmm_scan,          dim3(8),    dim3(256), 0, stream,
                       ws, out);
}
